// EF2GCNLayer_46273977647231
// MI455X (gfx1250) — matrix operand order repaired and checked by reading, not compiled, not run on hardware
//
#include <hip/hip_runtime.h>

// ---------------------------------------------------------------------------
// EF2GCN fused layer for MI455X (gfx1250, wave32, WMMA).
//   out = relu(mask * (X@W + (per-channel X@X)@W_prop + bias))
// Strategy: prep kernel transposes/converts X to channel-major bf16 (Xrow/Xcol
// in d_ws, 128 MB required), fused kernel does per-channel 256^3 GEMMs with
// v_wmma_f32_16x16x32_bf16 + fused f->o contraction + epilogue.
// LDS fills use gfx1250 async global->LDS copies (ASYNCcnt path), with a
// synchronous b128 fallback if the builtin is unavailable.
// ---------------------------------------------------------------------------

typedef unsigned short u16;
typedef __attribute__((ext_vector_type(16))) __bf16 v16bf;
typedef __attribute__((ext_vector_type(8)))  float  v8f;
typedef int v4ig __attribute__((vector_size(16)));   // matches builtin param

#define AS1 __attribute__((address_space(1)))
#define AS3 __attribute__((address_space(3)))

#if __has_builtin(__builtin_amdgcn_global_load_async_to_lds_b128)
  #define ASYNC_MODE 1
  #define ASYNC_CP16(gp, lp)                                                   \
    __builtin_amdgcn_global_load_async_to_lds_b128(                            \
        (AS1 v4ig*)(gp), (AS3 v4ig*)(lp), 0, 0)
#elif __has_builtin(__builtin_amdgcn_load_to_lds)
  #define ASYNC_MODE 2
  #define ASYNC_CP16(gp, lp)                                                   \
    __builtin_amdgcn_load_to_lds((void*)(gp), (AS3 void*)(lp), 16, 0, 0)
#else
  #define ASYNC_MODE 0
#endif

__device__ __forceinline__ void async_wait_all() {
#if ASYNC_MODE
 #if __has_builtin(__builtin_amdgcn_s_wait_asynccnt)
  __builtin_amdgcn_s_wait_asynccnt(0);
 #else
  asm volatile("s_wait_asynccnt 0" ::: "memory");
 #endif
#endif
}

#define NB 8
#define NN 256
#define NF 64
#define NO 64

__device__ __forceinline__ u16 f2bf(float x) {
  union { float f; unsigned int u; } v; v.f = x;
  unsigned int r = v.u + 0x7FFFu + ((v.u >> 16) & 1u);  // round-to-nearest-even
  return (u16)(r >> 16);
}

// ---------------------------------------------------------------------------
// Prep: X[b,i,j,f] (f32, f innermost) -> Xrow[(b*64+f)*256+i][k=j] bf16
//                                        Xcol[(b*64+f)*256+j][k=i] bf16
// One WG per (b, 16x16 ij tile); tile staged in LDS (stride-68 pad).
// ---------------------------------------------------------------------------
__global__ void __launch_bounds__(256) prep_kernel(const float* __restrict__ X,
                                                   u16* __restrict__ Xrow,
                                                   u16* __restrict__ Xcol) {
  extern __shared__ char smem[];
  float* tile = (float*)smem;                 // [ij:256][68] padded
  const int jt = blockIdx.x, it = blockIdx.y, b = blockIdx.z;
  const int i0 = it * 16, j0 = jt * 16;
  const int tid = threadIdx.x;

  const float4* X4 = (const float4*)X;
#pragma unroll
  for (int r = 0; r < 16; ++r) {
    int idx = r * 256 + tid;                  // 0..4095 float4s
    int fq = idx & 15;                        // 16 float4 per (i,j)
    int ij = idx >> 4;
    int i = ij >> 4, j = ij & 15;
    size_t g = (((size_t)b * NN + i0 + i) * NN + (j0 + j)) * (NF / 4) + fq;
#if ASYNC_MODE
    ASYNC_CP16(X4 + g, &tile[ij * 68 + fq * 4]);
#else
    *(float4*)&tile[ij * 68 + fq * 4] = X4[g];
#endif
  }
  async_wait_all();
  __syncthreads();

#pragma unroll
  for (int c = 0; c < 4; ++c) {
    int ch = tid * 4 + c;                     // 0..1023 chunks of 16
    int f = ch >> 4;
    int x = ch & 15;                          // i for Xrow, j for Xcol
    u16 hr[16], hc[16];
#pragma unroll
    for (int e = 0; e < 16; ++e) {
      hr[e] = f2bf(tile[(x * 16 + e) * 68 + f]);   // row i0+x, k=j0+e
      hc[e] = f2bf(tile[(e * 16 + x) * 68 + f]);   // row j0+x, k=i0+e
    }
    size_t rr = (((size_t)b * NF + f) * NN + (i0 + x)) * NN + j0;
    size_t rc = (((size_t)b * NF + f) * NN + (j0 + x)) * NN + i0;
    *(uint4*)&Xrow[rr]     = ((uint4*)hr)[0];
    *(uint4*)&Xrow[rr + 8] = ((uint4*)hr)[1];
    *(uint4*)&Xcol[rc]     = ((uint4*)hc)[0];
    *(uint4*)&Xcol[rc + 8] = ((uint4*)hc)[1];
  }
}

// ---------------------------------------------------------------------------
// Fused main kernel. One WG (8 waves) per (b, 16x16 ij tile).
// LDS map (bytes): [0,163840): phase A sA/sB, aliased in phase B by
// xt/xp/wt/wpt; [163840,164864): mask tile. Total 164864.
// ---------------------------------------------------------------------------
__global__ void __launch_bounds__(256) fused_kernel(const float* __restrict__ X,
                                                    const u16* __restrict__ Xrow,
                                                    const u16* __restrict__ Xcol,
                                                    const float* __restrict__ Mk,
                                                    const float* __restrict__ W,
                                                    const float* __restrict__ Wp,
                                                    const float* __restrict__ Wb,
                                                    float* __restrict__ out) {
  extern __shared__ char smem[];
  const int jt = blockIdx.x, it = blockIdx.y, b = blockIdx.z;
  const int i0 = it * 16, j0 = jt * 16;
  const int tid  = threadIdx.x;
  const int wave = tid >> 5, lane = tid & 31;
  const int ml = lane & 15, hl = lane >> 4;

  u16* sA = (u16*)smem;                       // [f:64][m:16][k:40 padded]
  u16* sB = sA + NF * 640;
  uint4* sA4 = (uint4*)sA;
  uint4* sB4 = (uint4*)sB;
  float* smask = (float*)(smem + 2 * NF * 640 * 2);   // +163840 B

  smask[tid] = Mk[((size_t)b * NN + i0 + (tid >> 4)) * NN + (j0 + (tid & 15))];

  // ---- Phase A: X_prop tile, per-channel GEMM, K accumulated over 8 chunks
  v8f accp[8];
#pragma unroll
  for (int c = 0; c < 8; ++c)
#pragma unroll
    for (int g = 0; g < 8; ++g) accp[c][g] = 0.0f;

  const uint4* Xr4 = (const uint4*)Xrow;      // 8 bf16 per uint4, 32 per k-row
  const uint4* Xc4 = (const uint4*)Xcol;

  for (int kc = 0; kc < 8; ++kc) {
    const int k0q = kc * 4;                   // uint4 offset along k
    __syncthreads();                          // previous chunk fully consumed
#pragma unroll
    for (int r = 0; r < 16; ++r) {
      int idx = r * 256 + tid;
      int kq = idx & 3, m = (idx >> 2) & 15, f = idx >> 6;
      size_t ga = (((size_t)b * NF + f) * NN + (i0 + m)) * 32 + k0q + kq;
      size_t gb = (((size_t)b * NF + f) * NN + (j0 + m)) * 32 + k0q + kq;
#if ASYNC_MODE
      ASYNC_CP16(Xr4 + ga, sA4 + f * 80 + m * 5 + kq);
      ASYNC_CP16(Xc4 + gb, sB4 + f * 80 + m * 5 + kq);
#else
      sA4[f * 80 + m * 5 + kq] = Xr4[ga];
      sB4[f * 80 + m * 5 + kq] = Xc4[gb];
#endif
    }
    async_wait_all();
    __syncthreads();
#pragma unroll
    for (int c = 0; c < 8; ++c) {
      int f = wave * 8 + c;
      // A frag (16x32 bf16): half0 K{0..7,16..23}, half1 K{8..15,24..31}
      const u16* pa = sA + f * 640 + ml * 40 + hl * 8;
      v16bf av;
      *(uint4*)&av       = *(const uint4*)pa;
      *((uint4*)&av + 1) = *(const uint4*)(pa + 16);
      // B frag (32x16 bf16): lane = column, half0 K{0..7,16..23}, half1 K{8..15,24..31}
      const u16* pb = sB + f * 640 + ml * 40 + hl * 8;
      v16bf bv;
      *(uint4*)&bv       = *(const uint4*)pb;
      *((uint4*)&bv + 1) = *(const uint4*)(pb + 16);
      accp[c] = __builtin_amdgcn_wmma_f32_16x16x32_bf16(
          false, av, false, bv, (short)0, accp[c], false, false);
    }
  }

  // ---- Phase B: (X_tile@W + Xp_tile@Wp + bias) * mask, relu
  __syncthreads();                            // phase A LDS reads done
  u16* xt  = (u16*)smem;                      // [r:256][f:72 padded] bf16
  u16* xp  = xt + 256 * 72;
  u16* wt  = xp + 256 * 72;                   // W^T  [o:64][f:72]
  u16* wpt = wt + 64 * 72;                    // Wp^T [o:64][f:72]

  // spill X_prop accumulators (C layout: M = g + 8*half, N = lane&15)
#pragma unroll
  for (int c = 0; c < 8; ++c) {
    int f = wave * 8 + c;
#pragma unroll
    for (int g = 0; g < 8; ++g) {
      int rr = (g + 8 * hl) * 16 + ml;
      xp[rr * 72 + f] = f2bf(accp[c][g]);
    }
  }
  // X tile rows (one per thread), converted to bf16
  {
    int r = tid;
    const float4* X4 = (const float4*)X;
    size_t gb = (((size_t)b * NN + i0 + (r >> 4)) * NN + (j0 + (r & 15))) * (NF / 4);
    u16 hh[64];
#pragma unroll
    for (int q = 0; q < 16; ++q) {
      float4 v = X4[gb + q];
      hh[q * 4 + 0] = f2bf(v.x); hh[q * 4 + 1] = f2bf(v.y);
      hh[q * 4 + 2] = f2bf(v.z); hh[q * 4 + 3] = f2bf(v.w);
    }
    uint4* dst = (uint4*)(xt + (size_t)r * 72);
#pragma unroll
    for (int q = 0; q < 8; ++q) dst[q] = ((uint4*)hh)[q];
  }
  // transposed bf16 weights
  {
    int o = tid >> 2, fq = tid & 3;
    u16 hw[16], hp[16];
#pragma unroll
    for (int e = 0; e < 16; ++e) {
      int f = fq * 16 + e;
      hw[e] = f2bf(W[f * NO + o]);
      hp[e] = f2bf(Wp[f * NO + o]);
    }
    *(uint4*)&wt[o * 72 + fq * 16]      = ((uint4*)hw)[0];
    *(uint4*)&wt[o * 72 + fq * 16 + 8]  = ((uint4*)hw)[1];
    *(uint4*)&wpt[o * 72 + fq * 16]     = ((uint4*)hp)[0];
    *(uint4*)&wpt[o * 72 + fq * 16 + 8] = ((uint4*)hp)[1];
  }
  __syncthreads();

  v8f acc[2][4];
#pragma unroll
  for (int mi = 0; mi < 2; ++mi)
#pragma unroll
    for (int nt = 0; nt < 4; ++nt) {
      float bvv = Wb[nt * 16 + ml];           // bias broadcast along rows
#pragma unroll
      for (int g = 0; g < 8; ++g) acc[mi][nt][g] = bvv;
    }

#pragma unroll
  for (int fc = 0; fc < 64; fc += 32) {
    v16bf ax[2], ap[2];
#pragma unroll
    for (int mi = 0; mi < 2; ++mi) {
      int row = (2 * wave + mi) * 16 + ml;
      const u16* p = xt + row * 72 + fc + hl * 8;
      *(uint4*)&ax[mi]       = *(const uint4*)p;
      *((uint4*)&ax[mi] + 1) = *(const uint4*)(p + 16);
      const u16* q = xp + row * 72 + fc + hl * 8;
      *(uint4*)&ap[mi]       = *(const uint4*)q;
      *((uint4*)&ap[mi] + 1) = *(const uint4*)(q + 16);
    }
    v16bf bw[4], bp[4];
#pragma unroll
    for (int nt = 0; nt < 4; ++nt) {
      int col = nt * 16 + ml;
      const u16* p = wt + col * 72 + fc + hl * 8;
      *(uint4*)&bw[nt]       = *(const uint4*)p;
      *((uint4*)&bw[nt] + 1) = *(const uint4*)(p + 16);
      const u16* q = wpt + col * 72 + fc + hl * 8;
      *(uint4*)&bp[nt]       = *(const uint4*)q;
      *((uint4*)&bp[nt] + 1) = *(const uint4*)(q + 16);
    }
#pragma unroll
    for (int mi = 0; mi < 2; ++mi)
#pragma unroll
      for (int nt = 0; nt < 4; ++nt) {
        acc[mi][nt] = __builtin_amdgcn_wmma_f32_16x16x32_bf16(
            false, ax[mi], false, bw[nt], (short)0, acc[mi][nt], false, false);
        acc[mi][nt] = __builtin_amdgcn_wmma_f32_16x16x32_bf16(
            false, ap[mi], false, bp[nt], (short)0, acc[mi][nt], false, false);
      }
  }

  // epilogue: mask, relu, coalesced f32 stores
#pragma unroll
  for (int mi = 0; mi < 2; ++mi) {
    int rbase = (2 * wave + mi) * 16;
#pragma unroll
    for (int g = 0; g < 8; ++g) {
      int r = rbase + g + 8 * hl;
      float mv = smask[r];
      int il = r >> 4, jl = r & 15;
      size_t ob = (((size_t)b * NN + i0 + il) * NN + (j0 + jl)) * NO;
#pragma unroll
      for (int nt = 0; nt < 4; ++nt) {
        float v = acc[mi][nt][g] * mv;
        out[ob + nt * 16 + ml] = v > 0.0f ? v : 0.0f;
      }
    }
  }
}

// ---------------------------------------------------------------------------
extern "C" void kernel_launch(void* const* d_in, const int* in_sizes, int n_in,
                              void* d_out, int out_size, void* d_ws, size_t ws_size,
                              hipStream_t stream) {
  (void)in_sizes; (void)n_in; (void)out_size; (void)ws_size;
  const float* X  = (const float*)d_in[0];
  const float* Mk = (const float*)d_in[1];
  // d_in[2] = n (unused by the reference computation)
  const float* W  = (const float*)d_in[3];
  const float* Wp = (const float*)d_in[4];
  const float* Wb = (const float*)d_in[5];
  float* out = (float*)d_out;

  // workspace: Xrow + Xcol, each B*F*N*N bf16 = 64 MB (needs ws_size >= 128 MB)
  u16* Xrow = (u16*)d_ws;
  u16* Xcol = Xrow + (size_t)NB * NF * NN * NN;

  const int ldsPrep  = 256 * 68 * 4;             // 69632 B
  const int ldsFused = 2 * NF * 640 * 2 + 1024;  // 164864 B
  (void)hipFuncSetAttribute((const void*)prep_kernel,
                            hipFuncAttributeMaxDynamicSharedMemorySize, ldsPrep);
  (void)hipFuncSetAttribute((const void*)fused_kernel,
                            hipFuncAttributeMaxDynamicSharedMemorySize, ldsFused);

  dim3 grid(NN / 16, NN / 16, NB), blk(256);
  prep_kernel<<<grid, blk, ldsPrep, stream>>>(X, Xrow, Xcol);
  fused_kernel<<<grid, blk, ldsFused, stream>>>(X, Xrow, Xcol, Mk, W, Wp, Wb, out);
}
